// HeteroGraphSAGE_28647431864642
// MI455X (gfx1250) — compile-verified, matrix operand order checked by reading, not run on hardware
//
#include <hip/hip_runtime.h>
#include <hip/hip_bf16.h>
#include <stdint.h>

#define N_U 50000
#define N_T 50000
#define NE  500000

typedef __attribute__((ext_vector_type(16))) __bf16 v16bf;
typedef __attribute__((ext_vector_type(8)))  float  v8f;

union FragAB { float4 f[2]; v16bf v; };

__device__ __forceinline__ unsigned short f2bf(float x) {
  unsigned int u = __float_as_uint(x);
  unsigned int r = (u + 0x7FFFu + ((u >> 16) & 1u)) >> 16;
  return (unsigned short)r;
}

// ---------------------------------------------------------------- fill -inf
__global__ __launch_bounds__(256) void fill_f32_kernel(float4* __restrict__ p,
                                                       float val, int n4) {
  int i = blockIdx.x * 256 + threadIdx.x;
  if (i < n4) p[i] = make_float4(val, val, val, val);
}

// ------------------------------------------- pack f32 [K,128] weights into
// bf16 arranged in per-lane WMMA B-fragment order: for (ktile,ntile), lane l
// holds 16 contiguous bf16 covering k = kt*32 + {0..7,16..23}(+8 if l>=16),
// column n = ntile*16 + (l&15).
__global__ __launch_bounds__(256) void pack_w_kernel(const float* __restrict__ W,
                                                     unsigned short* __restrict__ P,
                                                     int K) {
  int o = blockIdx.x * 256 + threadIdx.x;
  if (o >= K * 128) return;
  int e    = o & 15;
  int lane = (o >> 4) & 31;
  int flat = o >> 9;
  int ntile = flat & 7;
  int kt    = flat >> 3;
  int k = kt * 32 + ((e < 8) ? e : e + 8) + ((lane >= 16) ? 8 : 0);
  int n = ntile * 16 + (lane & 15);
  P[o] = f2bf(W[k * 128 + n]);
}

// ------------------------------------------------------------ edge scatter
// Native CDNA5 float atomic max (GLOBAL_ATOMIC_MAX_NUM_F32), device scope,
// non-returning (tracked on STOREcnt).
__device__ __forceinline__ void atomicMaxF(float* addr, float v) {
  asm volatile("global_atomic_max_num_f32 %0, %1, off scope:SCOPE_DEV"
               :: "v"(addr), "v"(v) : "memory");
}

__global__ __launch_bounds__(256) void scatter_max_kernel(
    const float* __restrict__ hsrc,
    const int* __restrict__ src,
    const int* __restrict__ dst,
    float* __restrict__ neigh,
    int nEdges, int d) {
  int tpe = d >> 2;              // float4 lanes per edge (32 or 64)
  int epb = 256 / tpe;           // edges per block
  int t = threadIdx.x;
  int e = blockIdx.x * epb + t / tpe;
  if (e >= nEdges) return;
  int c = (t % tpe) << 2;
  int s  = src[e];
  int dd = dst[e];
  const float4 v = *(const float4*)(hsrc + (size_t)s * d + c);
  float* tp = neigh + (size_t)dd * d + c;
  atomicMaxF(tp + 0, v.x);
  atomicMaxF(tp + 1, v.y);
  atomicMaxF(tp + 2, v.z);
  atomicMaxF(tp + 3, v.w);
}

// ---------------------------------------------------------- WMMA SAGE GEMM
// out[M,128] = combine( hdst@Wproj + bproj + relu?(concat(hdst,neigh)@Wfc + bfc) )
// Block: 80 rows x 128 cols, 8 waves; wave owns one 16-wide N-tile and 5
// 16-row M-subtiles; each B fragment is register-reused 5x, each A fragment
// feeds both the fc and proj WMMA where their K ranges overlap.
#define LDS_PAD 8
#define M_TILE  80
#define M_SUB   5
__global__ __launch_bounds__(256) void sage_wmma_kernel(
    const float* __restrict__ Adst,
    const float* __restrict__ Aneigh,
    int dDst, int dSrc,
    const unsigned short* __restrict__ Wfc,
    const float* __restrict__ bfc,
    const unsigned short* __restrict__ Wproj,
    const float* __restrict__ bproj,
    float* __restrict__ out,
    int doRelu, int combineMax) {
  __shared__ __align__(16) unsigned short Alds[M_TILE * (384 + LDS_PAD)];
  const int Ktot   = dDst + dSrc;
  const int stride = Ktot + LDS_PAD;
  const int row0   = blockIdx.x * M_TILE;

  // stage 80 rows of [hdst | fixed(neigh)] as bf16 into LDS (float4 in,
  // packed 4xbf16 out)
  const int K4 = Ktot >> 2;
  for (int i = threadIdx.x; i < M_TILE * K4; i += 256) {
    int r = i / K4;
    int c = (i - r * K4) << 2;
    float4 v;
    if (c < dDst) {
      v = *(const float4*)(Adst + (size_t)(row0 + r) * dDst + c);
    } else {
      v = *(const float4*)(Aneigh + (size_t)(row0 + r) * dSrc + (c - dDst));
      if (v.x == -__builtin_inff()) v.x = 0.0f;   // empty mailbox -> 0
      if (v.y == -__builtin_inff()) v.y = 0.0f;
      if (v.z == -__builtin_inff()) v.z = 0.0f;
      if (v.w == -__builtin_inff()) v.w = 0.0f;
    }
    uint2 pk;
    pk.x = (unsigned)f2bf(v.x) | ((unsigned)f2bf(v.y) << 16);
    pk.y = (unsigned)f2bf(v.z) | ((unsigned)f2bf(v.w) << 16);
    *(uint2*)(&Alds[r * stride + c]) = pk;
  }
  __syncthreads();

  const int w    = threadIdx.x >> 5;
  const int lane = threadIdx.x & 31;
  const int mloc = lane & 15;
  const int khi  = (lane >> 4) << 3;   // lanes 16..31 hold k+8 halves

  v8f accFc[M_SUB] = {};
  v8f accPr[M_SUB] = {};

  const int ktFc = Ktot >> 5;
  const int ktPr = dDst >> 5;
  for (int kt = 0; kt < ktFc; ++kt) {
    FragAB bF, bP;
    const float4* bp = (const float4*)(Wfc + ((((size_t)kt * 8 + w) * 32 + lane) << 4));
    bF.f[0] = bp[0];
    bF.f[1] = bp[1];
    const bool hasPr = (kt < ktPr);
    if (hasPr) {
      const float4* pp = (const float4*)(Wproj + ((((size_t)kt * 8 + w) * 32 + lane) << 4));
      bP.f[0] = pp[0];
      bP.f[1] = pp[1];
    }
#pragma unroll
    for (int mt = 0; mt < M_SUB; ++mt) {
      FragAB a;
      const float4* ap = (const float4*)(&Alds[(mt * 16 + mloc) * stride + (kt << 5) + khi]);
      a.f[0] = ap[0];          // k = base+0..7
      a.f[1] = ap[2];          // k = base+16..23
      accFc[mt] = __builtin_amdgcn_wmma_f32_16x16x32_bf16(false, a.v, false, bF.v,
                                                          (short)0, accFc[mt], false, false);
      if (hasPr)
        accPr[mt] = __builtin_amdgcn_wmma_f32_16x16x32_bf16(false, a.v, false, bP.v,
                                                            (short)0, accPr[mt], false, false);
    }
  }

  const int n     = (w << 4) + (lane & 15);
  const int mhalf = (lane >> 4) << 3;      // C layout: lanes>=16 hold M=8..15
  const float cbF = bfc[n];
  const float cbP = bproj[n];
#pragma unroll
  for (int mt = 0; mt < M_SUB; ++mt) {
#pragma unroll
    for (int r = 0; r < 8; ++r) {
      float v = accFc[mt][r] + cbF;
      if (doRelu) v = fmaxf(v, 0.0f);
      v += accPr[mt][r] + cbP;
      size_t idx = (size_t)(row0 + mt * 16 + mhalf + r) * 128 + n;
      if (combineMax) out[idx] = fmaxf(out[idx], v);
      else            out[idx] = v;
    }
  }
}

// -------------------------------------------------- layer-3 GEMV (d_out=1)
__global__ __launch_bounds__(256) void sage_gemv_kernel(
    const float* __restrict__ Adst,
    const float* __restrict__ Aneigh,
    int dDst, int dSrc,
    const float* __restrict__ wfc,
    const float* __restrict__ bfc,
    const float* __restrict__ wproj,
    const float* __restrict__ bproj,
    float* __restrict__ out,
    int M, int combineMax) {
  int row  = blockIdx.x * 8 + (threadIdx.x >> 5);
  int lane = threadIdx.x & 31;
  if (row >= M) return;
  float s = 0.0f;
  for (int j = lane; j < dDst; j += 32)
    s += Adst[(size_t)row * dDst + j] * (wfc[j] + wproj[j]);   // no relu in l3
  for (int j = lane; j < dSrc; j += 32) {
    float a = Aneigh[(size_t)row * dSrc + j];
    if (a == -__builtin_inff()) a = 0.0f;
    s += a * wfc[dDst + j];
  }
  for (int off = 16; off; off >>= 1) s += __shfl_xor(s, off, 32);
  if (lane == 0) {
    float v = s + bfc[0] + bproj[0];
    if (combineMax) out[row] = fmaxf(out[row], v);
    else            out[row] = v;
  }
}

// ======================================================================
extern "C" void kernel_launch(void* const* d_in, const int* in_sizes, int n_in,
                              void* d_out, int out_size, void* d_ws, size_t ws_size,
                              hipStream_t stream) {
  (void)n_in; (void)out_size; (void)ws_size;

  const float* feat_user  = (const float*)d_in[0];
  const float* feat_tweet = (const float*)d_in[1];

  // ---- map the 36 param leaves (indices 2..37). Relation ordering inside a
  // layer depends on dict flatten convention; detect via first leaf size:
  //   insertion order: uu first -> uu.fc.W = 256*128 = 32768
  //   sorted keys:     tu first -> tu.fc.W = 384*128 = 49152
  enum { UU = 0, UT = 1, TU = 2 };
  int order[3];
  if (in_sizes[2] == 32768) { order[0] = UU; order[1] = UT; order[2] = TU; }
  else                      { order[0] = TU; order[1] = UT; order[2] = UU; }

  const float *fcW[3][3], *fcB[3][3], *prW[3][3], *prB[3][3];
  int p = 2;
  for (int L = 0; L < 3; ++L)
    for (int k = 0; k < 3; ++k) {
      int r = order[k];
      fcW[L][r] = (const float*)d_in[p++];
      fcB[L][r] = (const float*)d_in[p++];
      prW[L][r] = (const float*)d_in[p++];
      prB[L][r] = (const float*)d_in[p++];
    }
  const int* src_uu = (const int*)d_in[38];
  const int* dst_uu = (const int*)d_in[39];
  const int* src_ut = (const int*)d_in[40];
  const int* dst_ut = (const int*)d_in[41];
  const int* src_tu = (const int*)d_in[42];
  const int* dst_tu = (const int*)d_in[43];

  // ---- workspace layout
  char* ws = (char*)d_ws;
  size_t off = 0;
  auto alloc = [&](size_t bytes) -> void* {
    void* r = ws + off;
    off = (off + bytes + 255) & ~(size_t)255;
    return r;
  };
  float* hu_a  = (float*)alloc((size_t)N_U * 128 * 4);
  float* hu_b  = (float*)alloc((size_t)N_U * 128 * 4);
  float* ht_a  = (float*)alloc((size_t)N_T * 128 * 4);
  float* ht_b  = (float*)alloc((size_t)N_T * 128 * 4);
  float* neigh = (float*)alloc((size_t)N_T * 256 * 4);

  // ---- pack layer-1/2 weights to bf16 fragment order
  static const int fcK[2][3] = {{256, 384, 384}, {256, 256, 256}};  // [L][UU,UT,TU]
  static const int prK[2][3] = {{128, 256, 128}, {128, 128, 128}};
  unsigned short* pfc[2][3];
  unsigned short* ppr[2][3];
  for (int L = 0; L < 2; ++L)
    for (int r = 0; r < 3; ++r) {
      pfc[L][r] = (unsigned short*)alloc((size_t)fcK[L][r] * 128 * 2);
      ppr[L][r] = (unsigned short*)alloc((size_t)prK[L][r] * 128 * 2);
      pack_w_kernel<<<(fcK[L][r] * 128 + 255) / 256, 256, 0, stream>>>(fcW[L][r], pfc[L][r], fcK[L][r]);
      pack_w_kernel<<<(prK[L][r] * 128 + 255) / 256, 256, 0, stream>>>(prW[L][r], ppr[L][r], prK[L][r]);
    }

  const float NEG_INF = -__builtin_inff();
  auto aggregate = [&](const float* hsrc, int dS, const int* src, const int* dst, int nDst) {
    int n4 = (nDst * dS) >> 2;
    fill_f32_kernel<<<(n4 + 255) / 256, 256, 0, stream>>>((float4*)neigh, NEG_INF, n4);
    int tpe = dS >> 2;
    int epb = 256 / tpe;
    scatter_max_kernel<<<(NE + epb - 1) / epb, 256, 0, stream>>>(hsrc, src, dst, neigh, NE, dS);
  };

  const int gU = N_U / M_TILE;   // 625
  const int gT = N_T / M_TILE;

  // ================= layer 1 (relu) : feats -> hu_a, ht_a =================
  aggregate(feat_tweet, 256, src_tu, dst_tu, N_U);                       // tu -> user
  sage_wmma_kernel<<<gU, 256, 0, stream>>>(feat_user, neigh, 128, 256,
      pfc[0][TU], fcB[0][TU], ppr[0][TU], prB[0][TU], hu_a, 1, 0);
  aggregate(feat_user, 128, src_uu, dst_uu, N_U);                        // uu -> user (max-combine)
  sage_wmma_kernel<<<gU, 256, 0, stream>>>(feat_user, neigh, 128, 128,
      pfc[0][UU], fcB[0][UU], ppr[0][UU], prB[0][UU], hu_a, 1, 1);
  aggregate(feat_user, 128, src_ut, dst_ut, N_T);                        // ut -> tweet
  sage_wmma_kernel<<<gT, 256, 0, stream>>>(feat_tweet, neigh, 256, 128,
      pfc[0][UT], fcB[0][UT], ppr[0][UT], prB[0][UT], ht_a, 1, 0);

  // ================= layer 2 (identity) : hu_a,ht_a -> hu_b,ht_b ==========
  aggregate(ht_a, 128, src_tu, dst_tu, N_U);
  sage_wmma_kernel<<<gU, 256, 0, stream>>>(hu_a, neigh, 128, 128,
      pfc[1][TU], fcB[1][TU], ppr[1][TU], prB[1][TU], hu_b, 0, 0);
  aggregate(hu_a, 128, src_uu, dst_uu, N_U);
  sage_wmma_kernel<<<gU, 256, 0, stream>>>(hu_a, neigh, 128, 128,
      pfc[1][UU], fcB[1][UU], ppr[1][UU], prB[1][UU], hu_b, 0, 1);
  aggregate(hu_a, 128, src_ut, dst_ut, N_T);
  sage_wmma_kernel<<<gT, 256, 0, stream>>>(ht_a, neigh, 128, 128,
      pfc[1][UT], fcB[1][UT], ppr[1][UT], prB[1][UT], ht_b, 0, 0);

  // ================= layer 3 (d_out = 1, GEMV) : -> d_out =================
  float* out_u = (float*)d_out;
  float* out_t = (float*)d_out + N_U;
  aggregate(ht_b, 128, src_tu, dst_tu, N_U);
  sage_gemv_kernel<<<(N_U + 7) / 8, 256, 0, stream>>>(hu_b, neigh, 128, 128,
      fcW[2][TU], fcB[2][TU], prW[2][TU], prB[2][TU], out_u, N_U, 0);
  aggregate(hu_b, 128, src_uu, dst_uu, N_U);
  sage_gemv_kernel<<<(N_U + 7) / 8, 256, 0, stream>>>(hu_b, neigh, 128, 128,
      fcW[2][UU], fcB[2][UU], prW[2][UU], prB[2][UU], out_u, N_U, 1);
  aggregate(hu_b, 128, src_ut, dst_ut, N_T);
  sage_gemv_kernel<<<(N_T + 7) / 8, 256, 0, stream>>>(ht_b, neigh, 128, 128,
      fcW[2][UT], fcB[2][UT], prW[2][UT], prB[2][UT], out_t, N_T, 0);
}
